// SimpleLSTM_37271726195399
// MI455X (gfx1250) — matrix operand order repaired and checked by reading, not compiled, not run on hardware
//
#include <hip/hip_runtime.h>

// Persistent masked-LSTM for MI455X (gfx1250, wave32, WMMA bf16).
// B=32, T=512, D=256, H=512. 32 workgroups, each owns 16 hidden units
// (all 4 gates -> 64 gate columns). Weights live in LDS as bf16 for the
// whole sequence; h is exchanged via a global bf16 double buffer using
// global_load_async_to_lds_b128, with a per-step arrival-counter grid
// barrier. The x-part of the GEMM (K=0..255) runs BEFORE the grid
// barrier to hide the sync latency; fragment loads are software-
// pipelined one K-step ahead of the WMMA chain.

typedef __attribute__((ext_vector_type(16))) __bf16 v16bf;
typedef __attribute__((ext_vector_type(8)))  float  v8f;

constexpr int BB   = 32;    // batch
constexpr int TT   = 512;   // time steps
constexpr int DD   = 256;   // input dim
constexpr int HH   = 512;   // hidden dim
constexpr int KT   = DD + HH;        // 768 combined K
constexpr int NWG  = HH / 16;        // 32 workgroups
constexpr int NTHR = 256;            // 8 waves

// LDS layout (bytes)
constexpr int OFF_W    = 0;                       // [64][768] bf16 = 98304
constexpr int OFF_A    = OFF_W + 64 * KT * 2;     // [32][768] bf16 = 49152
constexpr int OFF_G    = OFF_A + BB * KT * 2;     // [4][32][16] f32 = 8192
constexpr int OFF_C    = OFF_G + 4 * BB * 16 * 4; // [32][16] f32   = 2048
constexpr int OFF_BIAS = OFF_C + BB * 16 * 4;     // [64] f32       = 256
constexpr int LDS_BYTES = OFF_BIAS + 64 * 4;      // 157952

__device__ __forceinline__ unsigned short f2bf(float f) {
    unsigned int u = __float_as_uint(f);
    u += 0x7FFFu + ((u >> 16) & 1u);   // round-to-nearest-even
    return (unsigned short)(u >> 16);
}

#if __has_builtin(__builtin_amdgcn_cvt_pk_bf16_f32)
typedef __attribute__((ext_vector_type(2))) __bf16 v2bf;
__device__ __forceinline__ unsigned int pack2(float lo, float hi) {
    v2bf r = __builtin_amdgcn_cvt_pk_bf16_f32(lo, hi);   // v_cvt_pk_bf16_f32
    return __builtin_bit_cast(unsigned int, r);
}
#else
__device__ __forceinline__ unsigned int pack2(float lo, float hi) {
    return (unsigned int)f2bf(lo) | ((unsigned int)f2bf(hi) << 16);
}
#endif

// Branch-free activations: v_exp_f32 + v_rcp_f32, no ocml range reduction.
__device__ __forceinline__ float sigm(float v) {
    return __builtin_amdgcn_rcpf(1.0f + __expf(-v));
}
__device__ __forceinline__ float tanh_fast(float v) {
    // tanh(x) = 1 - 2/(1 + e^{2x})
    return __builtin_fmaf(-2.0f,
                          __builtin_amdgcn_rcpf(1.0f + __expf(2.0f * v)),
                          1.0f);
}

// Fragment loaders (ISA VGPR layouts, two ds_load_b128 each)
__device__ __forceinline__ v16bf ldA(const unsigned short* pa) {
    v16bf r;
    ((uint4*)&r)[0] = *(const uint4*)(pa);        // K = kbase .. kbase+7
    ((uint4*)&r)[1] = *(const uint4*)(pa + 16);   // K = kbase+16 .. +23
    return r;
}
__device__ __forceinline__ v16bf ldB(const unsigned short* pb) {
    v16bf r;
    ((uint4*)&r)[0] = *(const uint4*)(pb);        // K = kbase .. kbase+7
    ((uint4*)&r)[1] = *(const uint4*)(pb + 16);   // K = kbase+16 .. +23
    return r;
}

__global__ __launch_bounds__(NTHR, 1)
void lstm_persistent(const float* __restrict__ x,           // [B,T,D]
                     const unsigned char* __restrict__ msk,  // [B,T] (jax bool)
                     const float* __restrict__ c0,           // [B,H]
                     const float* __restrict__ h0,           // [B,H]
                     const float* __restrict__ init_c,       // [B,T,H]
                     const float* __restrict__ init_h,       // [B,T,H]
                     const float* __restrict__ Wi,           // [D,4H]
                     const float* __restrict__ Wh,           // [H,4H]
                     const float* __restrict__ bias,         // [4H]
                     float* __restrict__ out,                // [B,T,H]
                     float* __restrict__ cf,                 // [B,H]
                     float* __restrict__ hf,                 // [B,H]
                     int* __restrict__ bar,                  // [T] counters (pre-zeroed)
                     unsigned short* __restrict__ hbuf)      // [2][B][H] bf16
{
    extern __shared__ char smem[];
    unsigned short* wbuf  = (unsigned short*)(smem + OFF_W);
    unsigned short* abuf  = (unsigned short*)(smem + OFF_A);
    unsigned int*   abufw = (unsigned int*)abuf;
    float* gbuf  = (float*)(smem + OFF_G);
    float* cbuf  = (float*)(smem + OFF_C);
    float* biasL = (float*)(smem + OFF_BIAS);

    const int tid = threadIdx.x;
    const int wg  = blockIdx.x;       // 0..31
    const int U   = wg * 16;          // first hidden unit owned by this WG

    // ---- one-time: stage [Wi;Wh] slice into LDS as bf16, column-major ----
    for (int idx = tid; idx < 64 * KT; idx += NTHR) {
        const int col = idx & 63;          // local gate-column 0..63
        const int k   = idx >> 6;          // 0..767
        const int gcol = (col >> 4) * HH + U + (col & 15);   // global column
        const float v = (k < DD) ? Wi[(size_t)k * (4 * HH) + gcol]
                                 : Wh[(size_t)(k - DD) * (4 * HH) + gcol];
        wbuf[col * KT + k] = f2bf(v);
    }
    if (tid < 64) biasL[tid] = bias[(tid >> 4) * HH + U + (tid & 15)];
    for (int e = tid; e < BB * 16; e += NTHR) {
        const int bi = e >> 4, j = e & 15;
        cbuf[e] = c0[bi * HH + U + j];
    }
    __syncthreads();

    // ---- per-wave WMMA tile assignment ----
    const int lane = tid & 31;
    const int wv   = tid >> 5;          // 0..7
    const int mt   = wv & 1;            // M-tile (rows 16mt..16mt+15 of batch)
    const int nt   = wv >> 1;           // gate index 0..3 (N-tile)
    const int m16  = lane & 15;
    const int hi   = lane >> 4;         // half-wave
    const unsigned short* paBase = abuf + (mt * 16 + m16) * KT + hi * 8;
    const unsigned short* pbBase = wbuf + (nt * 16 + m16) * KT + hi * 8;

    // ---- register prefetch of x_0: 32 floats per thread ----
    float4 xr[8];
    #pragma unroll
    for (int q = 0; q < 4; ++q) {
        const int p  = tid + q * NTHR;          // 0..1023 (8-float chunks)
        const int bi = p >> 5;
        const int kp = (p & 31) << 3;
        const float4* src = (const float4*)(x + ((size_t)bi * TT + 0) * DD + kp);
        xr[2 * q]     = src[0];
        xr[2 * q + 1] = src[1];
    }

    for (int t = 0; t < TT; ++t) {
        // ---- stage x_t from registers into LDS (bf16) ----
        #pragma unroll
        for (int q = 0; q < 4; ++q) {
            const int p  = tid + q * NTHR;
            const int bi = p >> 5;
            const int kp = (p & 31) << 3;
            uint4 d;
            d.x = pack2(xr[2 * q].x,     xr[2 * q].y);
            d.y = pack2(xr[2 * q].z,     xr[2 * q].w);
            d.z = pack2(xr[2 * q + 1].x, xr[2 * q + 1].y);
            d.w = pack2(xr[2 * q + 1].z, xr[2 * q + 1].w);
            *(uint4*)(abuf + bi * KT + kp) = d;
        }
        __syncthreads();                     // x region of A visible

        // early mask loads for this step (consumed in elementwise tail)
        const int bi0 = tid >> 4;
        unsigned char mm[2];
        mm[0] = msk[bi0 * TT + t];
        mm[1] = msk[(bi0 + 16) * TT + t];

        // ---- x-part WMMA (K = 0..255), fragments pipelined 1 step ahead ----
        v8f acc = {};
        {
            v16bf A0 = ldA(paBase), B0 = ldB(pbBase);
            #pragma unroll
            for (int kk = 0; kk < DD / 32; ++kk) {
                v16bf A1 = A0, B1 = B0;
                if (kk + 1 < DD / 32) {
                    A1 = ldA(paBase + (kk + 1) * 32);
                    B1 = ldB(pbBase + (kk + 1) * 32);
                }
                acc = __builtin_amdgcn_wmma_f32_16x16x32_bf16(
                    false, A0, false, B0, (short)0, acc, false, false);
                A0 = A1; B0 = B1;
            }
        }

        // ---- grid barrier: wait until all WGs published h for step t-1 ----
        if (t > 0) {
            if ((tid & 31) == 0) {                   // one lane per wave spins
                while (__hip_atomic_load(&bar[t - 1], __ATOMIC_RELAXED,
                                         __HIP_MEMORY_SCOPE_AGENT) < NWG) {
                    __builtin_amdgcn_s_sleep(1);
                }
            }
            __threadfence();                          // acquire
        }

        // ---- stage h_{t-1} into LDS ----
        if (t == 0) {
            for (int p = tid; p < BB * (HH / 2); p += NTHR) {   // fp32 h0 -> bf16
                const int bi = p >> 8;
                const int kp = (p & 255) << 1;
                const float2 v = *(const float2*)(h0 + (size_t)bi * HH + kp);
                abufw[bi * (KT / 2) + (DD / 2) + (kp >> 1)] = pack2(v.x, v.y);
            }
        } else {
            // CDNA5 async copy: global bf16 h buffer -> LDS, no VGPR bounce
            const unsigned short* hsrc = hbuf + (size_t)((t - 1) & 1) * BB * HH;
            #pragma unroll
            for (int q = 0; q < 8; ++q) {
                const int p  = tid + q * NTHR;       // 0..2047 (16B chunks)
                const int bi = p >> 6;               // 64 chunks per batch row
                const int ck = p & 63;
                const unsigned gof = (unsigned)(bi * HH * 2 + ck * 16);
                const unsigned lof =
                    (unsigned)(size_t)(abuf + bi * KT + DD) + (unsigned)(ck * 16);
                asm volatile("global_load_async_to_lds_b128 %0, %1, %2"
                             :: "v"(lof), "v"(gof), "s"(hsrc) : "memory");
            }
            asm volatile("s_wait_asynccnt 0" ::: "memory");
        }
        __syncthreads();                     // h region of A visible

        // ---- prefetch x_{t+1}: loads hide behind the h-part WMMA chain ----
        if (t + 1 < TT) {
            #pragma unroll
            for (int q = 0; q < 4; ++q) {
                const int p  = tid + q * NTHR;
                const int bi = p >> 5;
                const int kp = (p & 31) << 3;
                const float4* src =
                    (const float4*)(x + ((size_t)bi * TT + (t + 1)) * DD + kp);
                xr[2 * q]     = src[0];
                xr[2 * q + 1] = src[1];
            }
        }

        // ---- h-part WMMA (K = 256..767), fragments pipelined 1 step ahead ----
        {
            v16bf A0 = ldA(paBase + (DD / 32) * 32), B0 = ldB(pbBase + (DD / 32) * 32);
            #pragma unroll
            for (int kk = DD / 32; kk < KT / 32; ++kk) {
                v16bf A1 = A0, B1 = B0;
                if (kk + 1 < KT / 32) {
                    A1 = ldA(paBase + (kk + 1) * 32);
                    B1 = ldB(pbBase + (kk + 1) * 32);
                }
                acc = __builtin_amdgcn_wmma_f32_16x16x32_bf16(
                    false, A0, false, B0, (short)0, acc, false, false);
                A0 = A1; B0 = B1;
            }
        }
        // scatter C tile to gate buffer: gbuf[gate][b][j]
        #pragma unroll
        for (int r = 0; r < 8; ++r) {
            const int bi = mt * 16 + hi * 8 + r;
            gbuf[nt * (BB * 16) + bi * 16 + m16] = acc[r];
        }
        __syncthreads();

        // ---- elementwise LSTM cell: 16 units x 32 batch, branch-free ----
        unsigned short* hdst = hbuf + (size_t)(t & 1) * BB * HH;
        #pragma unroll
        for (int q = 0; q < 2; ++q) {
            const int e  = tid + q * NTHR;     // 0..511
            const int bi = e >> 4;
            const int j  = e & 15;
            const float iv = gbuf[0 * 512 + e] + biasL[j];
            const float fv = gbuf[1 * 512 + e] + biasL[16 + j];
            const float gv = gbuf[2 * 512 + e] + biasL[32 + j];
            const float ov = gbuf[3 * 512 + e] + biasL[48 + j];
            const float cprev = cbuf[e];
            const float nc = sigm(fv) * cprev + sigm(iv) * tanh_fast(gv);
            const float nh = sigm(ov) * tanh_fast(nc);
            const bool  m  = mm[q] != 0;
            const size_t gix = ((size_t)bi * TT + t) * HH + U + j;
            const float cn = m ? nc : init_c[gix];
            const float hn = m ? nh : init_h[gix];
            cbuf[e] = cn;
            out[gix] = m ? nh : 0.0f;
            hdst[bi * HH + U + j] = f2bf(hn);
            if (t == TT - 1) {
                cf[bi * HH + U + j] = cn;
                hf[bi * HH + U + j] = hn;
            }
        }
        __threadfence();
        __syncthreads();
        if (tid == 0) atomicAdd(&bar[t], 1);   // publish step t
    }
}

extern "C" void kernel_launch(void* const* d_in, const int* in_sizes, int n_in,
                              void* d_out, int out_size, void* d_ws, size_t ws_size,
                              hipStream_t stream) {
    (void)in_sizes; (void)n_in; (void)out_size; (void)ws_size;
    const float*         x      = (const float*)d_in[0];
    const unsigned char* msk    = (const unsigned char*)d_in[1]; // jax bool -> 1B
    const float*         c0     = (const float*)d_in[2];
    const float*         h0     = (const float*)d_in[3];
    const float*         init_c = (const float*)d_in[4];
    const float*         init_h = (const float*)d_in[5];
    const float*         Wi     = (const float*)d_in[6];
    const float*         Wh     = (const float*)d_in[7];
    const float*         bias   = (const float*)d_in[8];

    float* out = (float*)d_out;                       // [B,T,H]
    float* cf  = out + (size_t)BB * TT * HH;          // [B,H]
    float* hf  = cf + (size_t)BB * HH;                // [B,H]

    int* bar = (int*)d_ws;                                          // T ints
    unsigned short* hbuf = (unsigned short*)((char*)d_ws + 4096);   // 2*B*H bf16

    (void)hipFuncSetAttribute((const void*)lstm_persistent,
                              hipFuncAttributeMaxDynamicSharedMemorySize,
                              LDS_BYTES);

    // zero arrival counters every call (graph-capture safe, deterministic)
    hipMemsetAsync(bar, 0, TT * sizeof(int), stream);

    lstm_persistent<<<dim3(NWG), dim3(NTHR), LDS_BYTES, stream>>>(
        x, msk, c0, h0, init_c, init_h, Wi, Wh, bias, out, cf, hf, bar, hbuf);
}
